// NormalizedCutLossChatGPT_58557584114371
// MI455X (gfx1250) — matrix operand order repaired and checked by reading, not compiled, not run on hardware
//
#include <hip/hip_runtime.h>

// ---------------------------------------------------------------------------
// Normalized-cut loss, MI455X (gfx1250, wave32, WMMA).
// One workgroup (256 thr = 8 waves) per 16x16 patch. Gram matrix via
// v_wmma_f32_16x16x32_bf16 using a bf16 hi/lo split of the fp32 features
// (3 WMMAs per 16x16 tile: hi*hi + hi*lo + lo*hi ~ fp32 accuracy).
// ---------------------------------------------------------------------------

typedef __attribute__((ext_vector_type(16))) __bf16 v16bf;
typedef __attribute__((ext_vector_type(8)))  __bf16 v8bf;
typedef __attribute__((ext_vector_type(8)))  float  v8f;

#define PSZ   16
#define P2    256      // ps*ps rows per patch
#define CCH   32       // feature channels == WMMA K
#define IMGW  256
#define IMGHW (256 * 256)

__device__ __forceinline__ float fast_exp2(float x) {
#if defined(__has_builtin)
#if __has_builtin(__builtin_amdgcn_exp2f)
  return __builtin_amdgcn_exp2f(x);   // v_exp_f32 (base-2)
#else
  return exp2f(x);
#endif
#else
  return exp2f(x);
#endif
}

__global__ __launch_bounds__(256)
void ncut_patch_kernel(const float* __restrict__ pred,   // (N,1,256,256)
                       const float* __restrict__ feat,   // (N,32,256,256)
                       float* __restrict__ patch_loss)   // (N*256)
{
  __shared__ __bf16 sh_hi[P2][CCH];   // 16 KB
  __shared__ __bf16 sh_lo[P2][CCH];   // 16 KB
  __shared__ float2 sh_sqp[P2];       // {||f||^2, p}  2 KB
  __shared__ float  sh_cut, sh_assoc;

  const int t     = threadIdx.x;
  const int patch = blockIdx.x;           // n*256 + l
  const int n     = patch >> 8;
  const int l     = patch & 255;
  const int hp    = l >> 4, wp = l & 15;
  const int py    = t >> 4, px = t & 15;
  const int h     = hp * PSZ + py;
  const int w     = wp * PSZ + px;

  if (t == 0) { sh_cut = 0.0f; sh_assoc = 0.0f; }

  // ---- stage: fp32 -> bf16 hi/lo split, row norms, predictions ----
  {
    const float* fbase = feat + (size_t)n * CCH * IMGHW + (size_t)h * IMGW + w;
    float sq = 0.0f;
#pragma unroll
    for (int c = 0; c < CCH; ++c) {
      float f = fbase[(size_t)c * IMGHW];
      sq = fmaf(f, f, sq);
      __bf16 hi = (__bf16)f;
      __bf16 lo = (__bf16)(f - (float)hi);
      sh_hi[t][c] = hi;
      sh_lo[t][c] = lo;
    }
    float p = pred[(size_t)n * IMGHW + (size_t)h * IMGW + w];
    sh_sqp[t] = make_float2(sq, p);
  }
  __syncthreads();

  // A = exp(-d_feat/(2*0.1^2) - d_sp/(2*16^2)),  d_sp = (dpy^2+dpx^2)/255^2
  const float C1L = -72.13475204444817f;                       // -log2e / 0.02
  const float C2L = -1.4426950408889634f / (512.0f * 65025.0f); // -log2e/(512*255^2)

  const int lane = t & 31;
  const int nl   = lane & 15;                  // N (column-in-tile) this lane holds
  const int off8 = (lane < 16) ? 0 : 8;        // M offset of the high lane half (C layout)
  const int aK0  = (lane < 16) ? 0 : 8;        // A-operand K base (runs at aK0, aK0+16)
  const int bK0  = (lane < 16) ? 0 : 8;        // B-operand K base (runs at bK0, bK0+16)
  const int wave = t >> 5;

  // spatial dx^2 per accumulator VGPR (M = v + off8, N = nl); dy is tile-constant
  float dx2[8];
#pragma unroll
  for (int v = 0; v < 8; ++v) {
    float dx = (float)(v + off8 - nl);
    dx2[v] = dx * dx;
  }

#pragma unroll
  for (int ii = 0; ii < 2; ++ii) {
    const int it = wave * 2 + ii;              // i-tile index (also py of rows)
    const int i0 = it * 16;
    const int ra = i0 + nl;                    // A row for this lane

    // A operands (ISA 16-bit A 16x32 layout): two contiguous 8-element runs
    v8bf ah0 = *(const v8bf*)&sh_hi[ra][aK0];
    v8bf ah1 = *(const v8bf*)&sh_hi[ra][aK0 + 16];
    v8bf al0 = *(const v8bf*)&sh_lo[ra][aK0];
    v8bf al1 = *(const v8bf*)&sh_lo[ra][aK0 + 16];
    v16bf a_hi, a_lo;
#pragma unroll
    for (int e = 0; e < 8; ++e) {
      a_hi[e] = ah0[e]; a_hi[e + 8] = ah1[e];
      a_lo[e] = al0[e]; a_lo[e + 8] = al1[e];
    }

    float sqi[8];
#pragma unroll
    for (int v = 0; v < 8; ++v) sqi[v] = sh_sqp[i0 + off8 + v].x;

    v8f accAp = {};                            // sum_j A_ij * p_j (partial over N-group)

    for (int jt = 0; jt < 16; ++jt) {
      const int rb = jt * 16 + nl;             // B column (= patch row j) for this lane

      // B operand (same K order as A): two contiguous 8-element runs
      v8bf bh0 = *(const v8bf*)&sh_hi[rb][bK0];
      v8bf bh1 = *(const v8bf*)&sh_hi[rb][bK0 + 16];
      v8bf bl0 = *(const v8bf*)&sh_lo[rb][bK0];
      v8bf bl1 = *(const v8bf*)&sh_lo[rb][bK0 + 16];
      v16bf b_hi, b_lo;
#pragma unroll
      for (int e = 0; e < 8; ++e) {
        b_hi[e] = bh0[e]; b_hi[e + 8] = bh1[e];
        b_lo[e] = bl0[e]; b_lo[e + 8] = bl1[e];
      }

      // inner = hi*hi + hi*lo + lo*hi  (3 chained bf16 WMMAs, K = C = 32)
      v8f c = {};
      c = __builtin_amdgcn_wmma_f32_16x16x32_bf16(false, a_lo, false, b_hi,
                                                  (short)0, c, false, false);
      c = __builtin_amdgcn_wmma_f32_16x16x32_bf16(false, a_hi, false, b_lo,
                                                  (short)0, c, false, false);
      c = __builtin_amdgcn_wmma_f32_16x16x32_bf16(false, a_hi, false, b_hi,
                                                  (short)0, c, false, false);

      const float2 sp = sh_sqp[rb];            // {sq_j, p_j}
      const float dy  = (float)(it - jt);
      const float dy2 = dy * dy;

#pragma unroll
      for (int v = 0; v < 8; ++v) {
        float d = fmaf(-2.0f, c[v], sqi[v] + sp.x);   // ||fi - fj||^2
        d = fmaxf(d, 0.0f);
        float e = fmaf(d, C1L, (dy2 + dx2[v]) * C2L); // base-2 exponent
        float A = fast_exp2(e);
        accAp[v] = fmaf(A, sp.y, accAp[v]);
      }
    }

    // reduce Ap over the 16-lane N group, then cut/assoc partials
    float cutP = 0.0f, assocP = 0.0f;
#pragma unroll
    for (int v = 0; v < 8; ++v) {
      float s = accAp[v];
      s += __shfl_xor(s, 1, 32);
      s += __shfl_xor(s, 2, 32);
      s += __shfl_xor(s, 4, 32);
      s += __shfl_xor(s, 8, 32);
      float pi = sh_sqp[i0 + off8 + v].y;
      assocP += s;
      cutP   += (1.0f - pi) * s;
    }
    if (nl == 0) {                              // one lane per 16-lane group
      atomicAdd(&sh_assoc, assocP);
      atomicAdd(&sh_cut, cutP);
    }
  }

  __syncthreads();
  if (t == 0) patch_loss[patch] = sh_cut / (sh_assoc + 1e-5f);
}

__global__ __launch_bounds__(256)
void ncut_reduce_kernel(const float* __restrict__ pl, float* __restrict__ out,
                        int count)
{
  __shared__ float sbuf[256];
  float s = 0.0f;
  for (int i = threadIdx.x; i < count; i += 256) s += pl[i];
  sbuf[threadIdx.x] = s;
  __syncthreads();
#pragma unroll
  for (int off = 128; off > 0; off >>= 1) {
    if (threadIdx.x < off) sbuf[threadIdx.x] += sbuf[threadIdx.x + off];
    __syncthreads();
  }
  if (threadIdx.x == 0) out[0] = sbuf[0] / (float)count;
}

extern "C" void kernel_launch(void* const* d_in, const int* in_sizes, int n_in,
                              void* d_out, int out_size, void* d_ws, size_t ws_size,
                              hipStream_t stream) {
  const float* pred = (const float*)d_in[0];   // (N,1,256,256) fp32
  const float* feat = (const float*)d_in[1];   // (N,32,256,256) fp32
  float* out = (float*)d_out;                  // scalar loss
  float* pl  = (float*)d_ws;                   // per-patch losses

  const int N  = in_sizes[0] / IMGHW;          // = 4
  const int NL = N * 256;                      // patches total

  ncut_patch_kernel<<<NL, 256, 0, stream>>>(pred, feat, pl);
  ncut_reduce_kernel<<<1, 256, 0, stream>>>(pl, out, NL);
}
